// ATTD_MambaBlock_87900800680721
// MI455X (gfx1250) — matrix operand order repaired and checked by reading, not compiled, not run on hardware
//
#include <hip/hip_runtime.h>
#include <hip/hip_bf16.h>
#include <math.h>

// ---------------- problem constants ----------------
#define D_MODEL  768
#define D_STATE  16
#define D_CONV   4
#define D_INNER  1536
#define BATCH    2
#define SEQLEN   2048
#define NROWS    (BATCH * SEQLEN)        // 4096 = GEMM M dim
#define XZ_N     (2 * D_INNER)           // 3072
#define XDBL_N   (1 + 2 * D_STATE)       // 33
#define XDBL_PAD 48                      // padded to 3 WMMA tiles
#define NCHUNK   32
#define CHLEN    (SEQLEN / NCHUNK)       // 64

typedef __attribute__((ext_vector_type(16))) __bf16 v16bf;
typedef __attribute__((ext_vector_type(8)))  float  v8f;

__device__ __forceinline__ float sigmoid_f(float x) { return 1.0f / (1.0f + __expf(-x)); }
__device__ __forceinline__ float softplus_f(float x) { return (x > 20.0f) ? x : log1pf(__expf(x)); }

__device__ __forceinline__ v16bf cvt_frag(float4 a, float4 b, float4 c, float4 d)
{
    v16bf v;
    v[0]  = (__bf16)a.x; v[1]  = (__bf16)a.y; v[2]  = (__bf16)a.z; v[3]  = (__bf16)a.w;
    v[4]  = (__bf16)b.x; v[5]  = (__bf16)b.y; v[6]  = (__bf16)b.z; v[7]  = (__bf16)b.w;
    v[8]  = (__bf16)c.x; v[9]  = (__bf16)c.y; v[10] = (__bf16)c.z; v[11] = (__bf16)c.w;
    v[12] = (__bf16)d.x; v[13] = (__bf16)d.y; v[14] = (__bf16)d.z; v[15] = (__bf16)d.w;
    return v;
}

// =====================================================================
// GEMM  C[M x ldC] = A[M x K] * W^T   (W is [Nw x K] row-major)
// One 16(M) x 64(N) strip per wave: A fragment loaded once per K-chunk,
// reused across 4 back-to-back v_wmma_f32_16x16x32_bf16. Branch-free
// inner loop (W row clamped, stores guarded outside). 4 waves / block.
// =====================================================================
__global__ void __launch_bounds__(128)
gemm_xt_wmma(const float* __restrict__ A, const float* __restrict__ W,
             float* __restrict__ C, int K, int Nw, int ldC, int Nquads)
{
    const int lane = threadIdx.x & 31;
    const int wave = threadIdx.x >> 5;
    const int quad = blockIdx.x * 4 + wave;       // 64-column strip index
    if (quad >= Nquads) return;
    const int tileM = blockIdx.y;
    const int half  = lane >> 4;                  // lane group 0/1
    const int r16   = lane & 15;

    const int mrow = tileM * 16 + r16;            // A row for this lane
    const float* Arow = A + (size_t)mrow * K;

    // W row per N-tile (clamped: out-of-range lanes produce unused columns)
    const float* Wrow[4];
    #pragma unroll
    for (int j = 0; j < 4; ++j) {
        int ncol = quad * 64 + j * 16 + r16;
        if (ncol > Nw - 1) ncol = Nw - 1;
        Wrow[j] = W + (size_t)ncol * K;
    }

    v8f acc0 = {0.f,0.f,0.f,0.f,0.f,0.f,0.f,0.f};
    v8f acc1 = acc0, acc2 = acc0, acc3 = acc0;

    for (int k0 = 0; k0 < K; k0 += 32) {
        // ---- A fragment: half 0 -> K {0..15}, half 1 -> K {16..31}
        const float* ap = Arow + k0 + half * 16;
        v16bf av = cvt_frag(*(const float4*)(ap +  0), *(const float4*)(ap +  4),
                            *(const float4*)(ap +  8), *(const float4*)(ap + 12));
        // ---- 4 B fragments, 4 WMMAs sharing av
        {
            const float* wp = Wrow[0] + k0 + half * 16;
            v16bf bv = cvt_frag(*(const float4*)(wp + 0), *(const float4*)(wp + 4),
                                *(const float4*)(wp + 8), *(const float4*)(wp + 12));
            acc0 = __builtin_amdgcn_wmma_f32_16x16x32_bf16(false, av, false, bv, (short)0, acc0, false, false);
        }
        {
            const float* wp = Wrow[1] + k0 + half * 16;
            v16bf bv = cvt_frag(*(const float4*)(wp + 0), *(const float4*)(wp + 4),
                                *(const float4*)(wp + 8), *(const float4*)(wp + 12));
            acc1 = __builtin_amdgcn_wmma_f32_16x16x32_bf16(false, av, false, bv, (short)0, acc1, false, false);
        }
        {
            const float* wp = Wrow[2] + k0 + half * 16;
            v16bf bv = cvt_frag(*(const float4*)(wp + 0), *(const float4*)(wp + 4),
                                *(const float4*)(wp + 8), *(const float4*)(wp + 12));
            acc2 = __builtin_amdgcn_wmma_f32_16x16x32_bf16(false, av, false, bv, (short)0, acc2, false, false);
        }
        {
            const float* wp = Wrow[3] + k0 + half * 16;
            v16bf bv = cvt_frag(*(const float4*)(wp + 0), *(const float4*)(wp + 4),
                                *(const float4*)(wp + 8), *(const float4*)(wp + 12));
            acc3 = __builtin_amdgcn_wmma_f32_16x16x32_bf16(false, av, false, bv, (short)0, acc3, false, false);
        }
    }

    // ---- store: acc element r -> (M = tileM*16 + r + 8*half, N = quad*64 + 16j + r16)
    #pragma unroll
    for (int j = 0; j < 4; ++j) {
        const v8f acc = (j == 0) ? acc0 : (j == 1) ? acc1 : (j == 2) ? acc2 : acc3;
        int n = quad * 64 + j * 16 + r16;
        if (n < ldC) {
            float* cp = C + (size_t)(tileM * 16 + half * 8) * ldC + n;
            #pragma unroll
            for (int r = 0; r < 8; ++r)
                cp[(size_t)r * ldC] = acc[r];
        }
    }
}

// =====================================================================
// Depthwise causal conv (k=4) + bias + SiLU.  lanes stride channel dim.
// =====================================================================
__global__ void conv_silu(const float* __restrict__ xz, const float* __restrict__ cw,
                          const float* __restrict__ cb, float* __restrict__ xc)
{
    int t = blockIdx.x * blockDim.x + threadIdx.x;
    if (t >= BATCH * SEQLEN * D_INNER) return;
    int d = t % D_INNER;
    int l = (t / D_INNER) % SEQLEN;
    int b = t / (D_INNER * SEQLEN);

    float acc = cb[d];
    #pragma unroll
    for (int j = 0; j < D_CONV; ++j) {
        int lj = l + j - (D_CONV - 1);
        if (lj >= 0)
            acc += cw[d * D_CONV + j] * xz[(size_t)(b * SEQLEN + lj) * XZ_N + d];
    }
    xc[(size_t)(b * SEQLEN + l) * D_INNER + d] = acc * sigmoid_f(acc);
}

// ---- vectorized read of one x_dbl row: dt scalar, B[16], C[16] ----
struct XRow { float dt; float B[D_STATE]; float C[D_STATE]; };

__device__ __forceinline__ XRow load_xrow(const float* __restrict__ xr)
{
    float4 q0 = *(const float4*)(xr +  0);
    float4 q1 = *(const float4*)(xr +  4);
    float4 q2 = *(const float4*)(xr +  8);
    float4 q3 = *(const float4*)(xr + 12);
    float4 q4 = *(const float4*)(xr + 16);
    float4 q5 = *(const float4*)(xr + 20);
    float4 q6 = *(const float4*)(xr + 24);
    float4 q7 = *(const float4*)(xr + 28);
    float  c15 = xr[32];
    XRow o;
    o.dt = q0.x;
    o.B[0]=q0.y;  o.B[1]=q0.z;  o.B[2]=q0.w;  o.B[3]=q1.x;
    o.B[4]=q1.y;  o.B[5]=q1.z;  o.B[6]=q1.w;  o.B[7]=q2.x;
    o.B[8]=q2.y;  o.B[9]=q2.z;  o.B[10]=q2.w; o.B[11]=q3.x;
    o.B[12]=q3.y; o.B[13]=q3.z; o.B[14]=q3.w; o.B[15]=q4.x;
    o.C[0]=q4.y;  o.C[1]=q4.z;  o.C[2]=q4.w;  o.C[3]=q5.x;
    o.C[4]=q5.y;  o.C[5]=q5.z;  o.C[6]=q5.w;  o.C[7]=q6.x;
    o.C[8]=q6.y;  o.C[9]=q6.z;  o.C[10]=q6.w; o.C[11]=q7.x;
    o.C[12]=q7.y; o.C[13]=q7.z; o.C[14]=q7.w; o.C[15]=c15;
    return o;
}

// =====================================================================
// Chunked selective scan, pass 1: per (b, d, chunk) compute
//   P[n] = prod dA over chunk,  S[n] = state after chunk (h_in = 0)
// =====================================================================
__global__ void scan_pass1(const float* __restrict__ xdbl, const float* __restrict__ xc,
                           const float* __restrict__ Wdt, const float* __restrict__ bdt,
                           const float* __restrict__ Alog,
                           float* __restrict__ Pbuf, float* __restrict__ Sbuf)
{
    int t = blockIdx.x * blockDim.x + threadIdx.x;
    if (t >= BATCH * NCHUNK * D_INNER) return;
    int d  = t % D_INNER;
    int ch = (t / D_INNER) % NCHUNK;
    int b  = t / (D_INNER * NCHUNK);

    const float wdt = Wdt[d], bd = bdt[d];
    float Arow[D_STATE], h[D_STATE], P[D_STATE];
    #pragma unroll
    for (int n = 0; n < D_STATE; ++n) {
        Arow[n] = -__expf(Alog[d * D_STATE + n]);
        h[n] = 0.f; P[n] = 1.f;
    }

    const int l0 = ch * CHLEN;
    for (int s = 0; s < CHLEN; ++s) {
        size_t row = (size_t)(b * SEQLEN + l0 + s);
        XRow xr = load_xrow(xdbl + row * XDBL_PAD);
        float delta = softplus_f(xr.dt * wdt + bd);
        float dxu   = delta * xc[row * D_INNER + d];
        #pragma unroll
        for (int n = 0; n < D_STATE; ++n) {
            float dA = __expf(delta * Arow[n]);
            h[n] = dA * h[n] + dxu * xr.B[n];
            P[n] *= dA;
        }
    }
    size_t base = (size_t)((b * NCHUNK + ch) * D_STATE) * D_INNER + d;
    #pragma unroll
    for (int n = 0; n < D_STATE; ++n) {
        Pbuf[base + (size_t)n * D_INNER] = P[n];
        Sbuf[base + (size_t)n * D_INNER] = h[n];
    }
}

// =====================================================================
// Pass 2: serial combine over the 32 chunks -> incoming state per chunk
// =====================================================================
__global__ void scan_pass2(const float* __restrict__ Pbuf, const float* __restrict__ Sbuf,
                           float* __restrict__ Hin)
{
    int t = blockIdx.x * blockDim.x + threadIdx.x;
    if (t >= BATCH * D_STATE * D_INNER) return;
    int d = t % D_INNER;
    int n = (t / D_INNER) % D_STATE;
    int b = t / (D_INNER * D_STATE);

    float h = 0.f;
    for (int ch = 0; ch < NCHUNK; ++ch) {
        size_t idx = (size_t)((b * NCHUNK + ch) * D_STATE + n) * D_INNER + d;
        Hin[idx] = h;
        h = Pbuf[idx] * h + Sbuf[idx];
    }
}

// =====================================================================
// Pass 3: replay each chunk from its true h_in, emit
//   y = sum_n h[n]*C[n] + x*D, gated by silu(z); overwrite xc with y.
// =====================================================================
__global__ void scan_pass3(const float* __restrict__ xdbl, const float* __restrict__ xz,
                           float* __restrict__ xc,
                           const float* __restrict__ Wdt, const float* __restrict__ bdt,
                           const float* __restrict__ Alog, const float* __restrict__ Dvec,
                           const float* __restrict__ Hin)
{
    int t = blockIdx.x * blockDim.x + threadIdx.x;
    if (t >= BATCH * NCHUNK * D_INNER) return;
    int d  = t % D_INNER;
    int ch = (t / D_INNER) % NCHUNK;
    int b  = t / (D_INNER * NCHUNK);

    const float wdt = Wdt[d], bd = bdt[d], Dd = Dvec[d];
    float Arow[D_STATE], h[D_STATE];
    size_t base = (size_t)((b * NCHUNK + ch) * D_STATE) * D_INNER + d;
    #pragma unroll
    for (int n = 0; n < D_STATE; ++n) {
        Arow[n] = -__expf(Alog[d * D_STATE + n]);
        h[n] = Hin[base + (size_t)n * D_INNER];
    }

    const int l0 = ch * CHLEN;
    for (int s = 0; s < CHLEN; ++s) {
        size_t row = (size_t)(b * SEQLEN + l0 + s);
        XRow xr = load_xrow(xdbl + row * XDBL_PAD);
        float delta = softplus_f(xr.dt * wdt + bd);
        float xv    = xc[row * D_INNER + d];
        float dxu   = delta * xv;
        float y = 0.f;
        #pragma unroll
        for (int n = 0; n < D_STATE; ++n) {
            float dA = __expf(delta * Arow[n]);
            h[n] = dA * h[n] + dxu * xr.B[n];
            y += h[n] * xr.C[n];
        }
        y += xv * Dd;
        float z = xz[row * XZ_N + D_INNER + d];
        y *= z * sigmoid_f(z);
        xc[row * D_INNER + d] = y;   // overwrite conv buffer with gated y
    }
}

// =====================================================================
extern "C" void kernel_launch(void* const* d_in, const int* in_sizes, int n_in,
                              void* d_out, int out_size, void* d_ws, size_t ws_size,
                              hipStream_t stream)
{
    const float* hidden = (const float*)d_in[0];
    const float* W_in   = (const float*)d_in[1];
    const float* conv_w = (const float*)d_in[2];
    const float* conv_b = (const float*)d_in[3];
    const float* W_x    = (const float*)d_in[4];
    const float* W_dt   = (const float*)d_in[5];
    const float* b_dt   = (const float*)d_in[6];
    const float* A_log  = (const float*)d_in[7];
    const float* Dvec   = (const float*)d_in[8];
    const float* W_out  = (const float*)d_in[9];
    float* out = (float*)d_out;

    char* p = (char*)d_ws;
    float* xz   = (float*)p; p += (size_t)NROWS * XZ_N * sizeof(float);       // 50.3 MB
    float* xc   = (float*)p; p += (size_t)NROWS * D_INNER * sizeof(float);    // 25.2 MB (conv x, then gated y)
    float* xdbl = (float*)p; p += (size_t)NROWS * XDBL_PAD * sizeof(float);   // 0.8 MB
    float* Pbuf = (float*)p; p += (size_t)BATCH * NCHUNK * D_STATE * D_INNER * sizeof(float); // 6.3 MB
    float* Sbuf = (float*)p; p += (size_t)BATCH * NCHUNK * D_STATE * D_INNER * sizeof(float); // 6.3 MB
    float* Hin  = (float*)p;                                                  // 6.3 MB

    // 1) in_proj: xz = hidden @ W_in^T      (M=4096, K=768, N=3072 -> 48 quads)
    gemm_xt_wmma<<<dim3(12, NROWS / 16), 128, 0, stream>>>(
        hidden, W_in, xz, D_MODEL, XZ_N, XZ_N, XZ_N / 64);

    // 2) depthwise causal conv + SiLU on x half
    int nconv = BATCH * SEQLEN * D_INNER;
    conv_silu<<<(nconv + 255) / 256, 256, 0, stream>>>(xz, conv_w, conv_b, xc);

    // 3) x_proj: xdbl = x_conv @ W_x^T      (M=4096, K=1536, N=33 pad 48 -> 1 quad)
    gemm_xt_wmma<<<dim3(1, NROWS / 16), 128, 0, stream>>>(
        xc, W_x, xdbl, D_INNER, XDBL_N, XDBL_PAD, 1);

    // 4) chunked selective scan
    int n1 = BATCH * NCHUNK * D_INNER;
    scan_pass1<<<(n1 + 255) / 256, 256, 0, stream>>>(xdbl, xc, W_dt, b_dt, A_log, Pbuf, Sbuf);
    int n2 = BATCH * D_STATE * D_INNER;
    scan_pass2<<<(n2 + 255) / 256, 256, 0, stream>>>(Pbuf, Sbuf, Hin);
    scan_pass3<<<(n1 + 255) / 256, 256, 0, stream>>>(xdbl, xz, xc, W_dt, b_dt, A_log, Dvec, Hin);

    // 5) out_proj: out = y @ W_out^T        (M=4096, K=1536, N=768 -> 12 quads)
    gemm_xt_wmma<<<dim3(3, NROWS / 16), 128, 0, stream>>>(
        xc, W_out, out, D_INNER, D_MODEL, D_MODEL, D_MODEL / 64);
}
